// Contrast_76270029243144
// MI455X (gfx1250) — matrix operand order repaired and checked by reading, not compiled, not run on hardware
//
#include <hip/hip_runtime.h>
#include <hip/hip_bf16.h>
#include <math.h>

// ---------------------------------------------------------------------------
// Contrastive loss, fused for MI455X (gfx1250, wave32, WMMA).
//
// Roofline: 275 GFLOP of 128x128 GEMMs over 2.1M rows vs 2 GiB of fp32
// sample reads. fp32 VALU would be ~3 ms (compute-bound); bf16 WMMA with
// f32 accumulation brings compute to ~HBM time (~100 us). So: convert
// activations/weights to bf16 on the fly, do all matmuls with
// v_wmma_f32_16x16x32_bf16, keep everything else (ELU, cos, exp, log) in f32.
// ---------------------------------------------------------------------------

typedef __attribute__((ext_vector_type(16))) __bf16 v16bf;
typedef __attribute__((ext_vector_type(8)))  __bf16 v8bf;
typedef __attribute__((ext_vector_type(8)))  float  v8f;

#define LDSTR 136            // padded LDS row stride in bf16 halves (breaks bank conflicts)
#define NROWS 8192
#define SSAMP 256
#define DDIM  128

// LDS layout offsets (bytes)
#define OFF_W1   0            // 128*136*2 = 34816
#define OFF_W2   34816
#define OFF_X    69632        // 64*136*2 = 17408
#define OFF_H    87040
#define OFF_B1   104448       // 128 f32
#define OFF_B2   104960
#define OFF_TM   105472       // 128 f32
#define OFF_DOT  105984       // 64 f32
#define OFF_NRM  106240       // 64 f32
#define OFF_TNSQ 106496       // 1 f32
#define SMEM_BYTES 106752

__device__ __forceinline__ v8f wmma_bf16(v16bf a, v16bf b, v8f c) {
    // emits v_wmma_f32_16x16x32_bf16
    return __builtin_amdgcn_wmma_f32_16x16x32_bf16(false, a, false, b,
                                                   (short)0, c, false, false);
}

// A fragment (16x32 bf16, ISA 7.12.2): lane L holds row M=L%16.
// lanes 0-15: K = kb+{0..7} in v0-3, kb+{16..23} in v4-7
// lanes 16-31: K = kb+{8..15},{24..31}
__device__ __forceinline__ v16bf load_a_frag(const __bf16* base, int rowbase,
                                             int kb, int lane) {
    const int m    = lane & 15;
    const int koff = (lane & 16) ? 8 : 0;
    const __bf16* p = base + (rowbase + m) * LDSTR + kb + koff;
    v8bf lo = *(const v8bf*)p;          // ds_load_b128
    v8bf hi = *(const v8bf*)(p + 16);   // ds_load_b128
    v16bf a;
#pragma unroll
    for (int i = 0; i < 8; ++i) { a[i] = lo[i]; a[i + 8] = hi[i]; }
    return a;
}

// B fragment (32x16 bf16): B[k][n] = W[n][k]  (computing X @ W^T).
// lane L holds column n=L%16; lanes 0-15 cover K=kb+{0..7},{16..23}, lanes 16-31 K=kb+{8..15},{24..31},
// halves of W row n (same K order as A fragment).
__device__ __forceinline__ v16bf load_b_frag(const __bf16* w, int nb,
                                             int kb, int lane) {
    const int nn   = lane & 15;
    const int koff = (lane & 16) ? 8 : 0;
    const __bf16* p = w + (nb + nn) * LDSTR + kb + koff;
    v8bf lo = *(const v8bf*)p;
    v8bf hi = *(const v8bf*)(p + 16);
    v16bf b;
#pragma unroll
    for (int i = 0; i < 8; ++i) { b[i] = lo[i]; b[i + 8] = hi[i]; }
    return b;
}

__device__ __forceinline__ float elu_f(float x) {
    return x > 0.0f ? x : (__expf(x) - 1.0f);
}

// fp32 -> bf16 conversion of the 4 weight matrices into workspace
__global__ __launch_bounds__(256)
void prep_weights(const float* __restrict__ wi1, const float* __restrict__ wi2,
                  const float* __restrict__ wt1, const float* __restrict__ wt2,
                  __bf16* __restrict__ out) {
    int i   = blockIdx.x * 256 + threadIdx.x;   // 0 .. 65535
    int m   = i >> 14;
    int idx = i & 16383;
    const float* src = (m == 0) ? wi1 : (m == 1) ? wi2 : (m == 2) ? wt1 : wt2;
    out[i] = (__bf16)src[idx];
}

// One block per n. 8 waves: wave = (row_tile rt = wid&3) x (col half ch = wid>>2).
__global__ __launch_bounds__(256)
void branch_kernel(const float* __restrict__ target,   // [N,128]
                   const float* __restrict__ samples,  // [N,256,128]
                   const float* __restrict__ mask,     // [N,256]
                   const __bf16* __restrict__ wb1,     // [128,128] bf16 (row n, col k)
                   const __bf16* __restrict__ wb2,
                   const float* __restrict__ b1g,
                   const float* __restrict__ b2g,
                   float* __restrict__ loss_out)       // [N]
{
    extern __shared__ char smem[];
    __bf16* w1s   = (__bf16*)(smem + OFF_W1);
    __bf16* w2s   = (__bf16*)(smem + OFF_W2);
    __bf16* xs    = (__bf16*)(smem + OFF_X);
    __bf16* hs    = (__bf16*)(smem + OFF_H);
    float*  b1s   = (float*)(smem + OFF_B1);
    float*  b2s   = (float*)(smem + OFF_B2);
    float*  tm_sh = (float*)(smem + OFF_TM);
    float*  dot_sh= (float*)(smem + OFF_DOT);
    float*  nrm_sh= (float*)(smem + OFF_NRM);
    float*  tnsq  = (float*)(smem + OFF_TNSQ);

    const int tid  = threadIdx.x;
    const int lane = tid & 31;
    const int wid  = tid >> 5;
    const int n    = blockIdx.x;

    // ---- prologue: weights (bf16) + biases into LDS -----------------------
    // u32 copy, unpadded src row (64 u32) -> padded LDS row (68 u32)
    for (int idx = tid; idx < 128 * 64; idx += 256) {
        int r = idx >> 6, c = idx & 63;
        ((unsigned*)w1s)[r * 68 + c] = ((const unsigned*)wb1)[idx];
        ((unsigned*)w2s)[r * 68 + c] = ((const unsigned*)wb2)[idx];
    }
    if (tid < 128) { b1s[tid] = b1g[tid]; b2s[tid] = b2g[tid]; }
    if (tid == 0)  *tnsq = 0.0f;
    // zero rows 1..15 of X and H (target stage uses only row 0)
    for (int idx = LDSTR + tid; idx < 16 * LDSTR; idx += 256) {
        xs[idx] = (__bf16)0.0f;
        hs[idx] = (__bf16)0.0f;
    }
    // target row n -> X row 0 (bf16)
    if (tid < 32) {
        float4 v = ((const float4*)(target + (size_t)n * DDIM))[tid];
        __bf16* d = xs + tid * 4;
        d[0] = (__bf16)v.x; d[1] = (__bf16)v.y; d[2] = (__bf16)v.z; d[3] = (__bf16)v.w;
    }
    __syncthreads();

    // ---- target MLP: tm = W2 @ ELU(W1 @ t + b1) + b2, one 16-row tile -----
    {
        const int nc = wid;                       // each wave: one 16-col tile
        v8f acc = {0.f,0.f,0.f,0.f,0.f,0.f,0.f,0.f};
#pragma unroll
        for (int kc = 0; kc < 4; ++kc) {
            v16bf a = load_a_frag(xs, 0, kc * 32, lane);
            v16bf b = load_b_frag(w1s, nc * 16, kc * 32, lane);
            acc = wmma_bf16(a, b, acc);
        }
        if (lane < 16) {                          // row m=0 lives in v0, lanes 0-15
            int col = nc * 16 + lane;
            hs[col] = (__bf16)elu_f(acc[0] + b1s[col]);
        }
    }
    __syncthreads();
    {
        const int nc = wid;
        v8f acc = {0.f,0.f,0.f,0.f,0.f,0.f,0.f,0.f};
#pragma unroll
        for (int kc = 0; kc < 4; ++kc) {
            v16bf a = load_a_frag(hs, 0, kc * 32, lane);
            v16bf b = load_b_frag(w2s, nc * 16, kc * 32, lane);
            acc = wmma_bf16(a, b, acc);
        }
        if (lane < 16) {
            int col = nc * 16 + lane;
            float y = acc[0] + b2s[col];
            tm_sh[col] = y;
            atomicAdd(tnsq, y * y);               // ds_add_f32
        }
    }
    __syncthreads();

    // ---- per-lane constants for the sample loop ---------------------------
    const int rt = wid & 3;                        // row tile (16 rows) in 64-row chunk
    const int ch = wid >> 2;                       // column half: 0 -> cols 0-63, 1 -> 64-127
    float b1v[4], b2v[4], tmv[4];
#pragma unroll
    for (int ct = 0; ct < 4; ++ct) {
        int col = (ch * 4 + ct) * 16 + (lane & 15);
        b1v[ct] = b1s[col]; b2v[ct] = b2s[col]; tmv[ct] = tm_sh[col];
    }
    const float tn = sqrtf(*tnsq);

    float acc_all = 0.0f, acc_pos = 0.0f;

    // ---- 4 chunks of 64 sample rows --------------------------------------
    for (int c = 0; c < 4; ++c) {
        // stage X chunk fp32 -> bf16 LDS (coalesced float4 loads)
        const float4* gsrc =
            (const float4*)(samples + ((size_t)n * SSAMP + (size_t)c * 64) * DDIM);
        for (int idx = tid; idx < 2048; idx += 256) {
            int r = idx >> 5, c4 = idx & 31;
            float4 v = gsrc[idx];
            __bf16* d = xs + r * LDSTR + c4 * 4;
            d[0] = (__bf16)v.x; d[1] = (__bf16)v.y;
            d[2] = (__bf16)v.z; d[3] = (__bf16)v.w;
        }
        if (tid < 64) { dot_sh[tid] = 0.0f; nrm_sh[tid] = 0.0f; }
        __syncthreads();

        // GEMM1: H = ELU(X @ W1^T + b1)
        {
            v8f acc[4];
#pragma unroll
            for (int ct = 0; ct < 4; ++ct)
                acc[ct] = (v8f){0.f,0.f,0.f,0.f,0.f,0.f,0.f,0.f};
#pragma unroll
            for (int kc = 0; kc < 4; ++kc) {
                v16bf a = load_a_frag(xs, rt * 16, kc * 32, lane);
#pragma unroll
                for (int ct = 0; ct < 4; ++ct) {
                    v16bf b = load_b_frag(w1s, (ch * 4 + ct) * 16, kc * 32, lane);
                    acc[ct] = wmma_bf16(a, b, acc[ct]);
                }
            }
            const int rbase = rt * 16 + ((lane & 16) ? 8 : 0);
#pragma unroll
            for (int ct = 0; ct < 4; ++ct) {
                int col = (ch * 4 + ct) * 16 + (lane & 15);
#pragma unroll
                for (int v = 0; v < 8; ++v) {
                    float h = elu_f(acc[ct][v] + b1v[ct]);
                    hs[(rbase + v) * LDSTR + col] = (__bf16)h;
                }
            }
        }
        __syncthreads();

        // GEMM2: Y = H @ W2^T + b2, fused dot(tm, y) and ||y||^2
        {
            v8f acc[4];
#pragma unroll
            for (int ct = 0; ct < 4; ++ct)
                acc[ct] = (v8f){0.f,0.f,0.f,0.f,0.f,0.f,0.f,0.f};
#pragma unroll
            for (int kc = 0; kc < 4; ++kc) {
                v16bf a = load_a_frag(hs, rt * 16, kc * 32, lane);
#pragma unroll
                for (int ct = 0; ct < 4; ++ct) {
                    v16bf b = load_b_frag(w2s, (ch * 4 + ct) * 16, kc * 32, lane);
                    acc[ct] = wmma_bf16(a, b, acc[ct]);
                }
            }
            float dotp[8], nrq[8];
#pragma unroll
            for (int v = 0; v < 8; ++v) { dotp[v] = 0.0f; nrq[v] = 0.0f; }
#pragma unroll
            for (int ct = 0; ct < 4; ++ct) {
#pragma unroll
                for (int v = 0; v < 8; ++v) {
                    float y = acc[ct][v] + b2v[ct];
                    dotp[v] += y * tmv[ct];
                    nrq[v]  += y * y;
                }
            }
            // reduce over the 16 lanes of each half-wave (n dimension)
#pragma unroll
            for (int v = 0; v < 8; ++v) {
                float d = dotp[v], q = nrq[v];
#pragma unroll
                for (int m = 1; m < 16; m <<= 1) {
                    d += __shfl_xor(d, m, 32);
                    q += __shfl_xor(q, m, 32);
                }
                if ((lane & 15) == 0) {
                    int row = rt * 16 + v + ((lane & 16) ? 8 : 0);
                    atomicAdd(&dot_sh[row], d);   // combines ch=0/ch=1 halves
                    atomicAdd(&nrm_sh[row], q);
                }
            }
        }
        __syncthreads();

        // per-row similarity: sim = exp(cos / tau)
        if (tid < 64) {
            float sn   = sqrtf(nrm_sh[tid]);
            float cosv = dot_sh[tid] / (tn * sn);
            float sim  = __expf(cosv * (1.0f / 0.8f));       // 1/TAU
            float mk   = mask[(size_t)n * SSAMP + (size_t)c * 64 + tid];
            acc_all += sim;
            acc_pos += sim * mk;
        }
        __syncthreads();
    }

    // ---- block reduction of Sum(sim), Sum(sim*mask) -----------------------
    if (tid < 64) { dot_sh[tid] = acc_all; nrm_sh[tid] = acc_pos; }
    __syncthreads();
    if (tid == 0) {
        float sa = 0.0f, sp = 0.0f;
        for (int i = 0; i < 64; ++i) { sa += dot_sh[i]; sp += nrm_sh[i]; }
        // loss = -log( sum(sim*mask) / (sum(sim)+eps) )
        loss_out[n] = logf(sa + 1e-8f) - logf(sp);
    }
}

__global__ __launch_bounds__(256)
void reduce_kernel(const float* __restrict__ li, const float* __restrict__ lt,
                   float* __restrict__ out) {
    __shared__ float sa[256], sb[256];
    float a = 0.0f, b = 0.0f;
    for (int i = threadIdx.x; i < NROWS; i += 256) { a += li[i]; b += lt[i]; }
    sa[threadIdx.x] = a; sb[threadIdx.x] = b;
    __syncthreads();
    for (int s = 128; s > 0; s >>= 1) {
        if (threadIdx.x < (unsigned)s) {
            sa[threadIdx.x] += sa[threadIdx.x + s];
            sb[threadIdx.x] += sb[threadIdx.x + s];
        }
        __syncthreads();
    }
    if (threadIdx.x == 0) {
        float mi = sa[0] / (float)NROWS;
        float mt = sb[0] / (float)NROWS;
        out[0] = 0.5f * mi + 0.5f * mt;   // LAM = 0.5
        out[1] = mi;
        out[2] = mt;
    }
}

extern "C" void kernel_launch(void* const* d_in, const int* in_sizes, int n_in,
                              void* d_out, int out_size, void* d_ws, size_t ws_size,
                              hipStream_t stream) {
    const float* t_inter = (const float*)d_in[0];
    const float* t_intra = (const float*)d_in[1];
    const float* s_inter = (const float*)d_in[2];
    const float* s_intra = (const float*)d_in[3];
    const float* m_inter = (const float*)d_in[4];
    const float* m_intra = (const float*)d_in[5];
    const float* Wi1 = (const float*)d_in[6];  const float* bi1 = (const float*)d_in[7];
    const float* Wi2 = (const float*)d_in[8];  const float* bi2 = (const float*)d_in[9];
    const float* Wt1 = (const float*)d_in[10]; const float* bt1 = (const float*)d_in[11];
    const float* Wt2 = (const float*)d_in[12]; const float* bt2 = (const float*)d_in[13];

    char* ws = (char*)d_ws;
    __bf16* wb     = (__bf16*)ws;                 // 4 x 16384 bf16 = 128 KB
    float*  loss_i = (float*)(ws + 131072);       // 8192 f32
    float*  loss_t = (float*)(ws + 163840);       // 8192 f32

    prep_weights<<<256, 256, 0, stream>>>(Wi1, Wi2, Wt1, Wt2, wb);

    branch_kernel<<<NROWS, 256, SMEM_BYTES, stream>>>(
        t_inter, s_inter, m_inter, wb, wb + 16384, bi1, bi2, loss_i);
    branch_kernel<<<NROWS, 256, SMEM_BYTES, stream>>>(
        t_intra, s_intra, m_intra, wb + 32768, wb + 49152, bt1, bt2, loss_t);

    reduce_kernel<<<1, 256, 0, stream>>>(loss_i, loss_t, (float*)d_out);
}
